// BioMoR_RNN_39548058861849
// MI455X (gfx1250) — matrix operand order repaired and checked by reading, not compiled, not run on hardware
//
#include <hip/hip_runtime.h>
#include <math.h>

// Problem constants (from reference)
#define HIDDEN 128
#define G3     384      // 3*H gate width (r,z,n)
#define CTXN   6
#define TSTEPS 1024
#define OUTN   4

// Tiling
#define TB     16       // batch rows per workgroup
#define NTH    256      // 8 wave32 waves
#define HWS    388      // padded row stride for combined pre-activation buffer
#define KAUG   160      // augmented K: 128 (h) + 6 (x) + 26 zero pad -> 5 chunks of 32
#define XWNS   132      // padded row stride for xw_n buffer
#define PHS    20       // padded row stride for head / reflex tiles

typedef __attribute__((ext_vector_type(16))) __bf16 v16bf;
typedef __attribute__((ext_vector_type(8)))  __bf16 v8bf;
typedef __attribute__((ext_vector_type(8)))  float  v8f;
typedef __attribute__((ext_vector_type(4)))  float  v4f;

__device__ __forceinline__ float fast_tanh(float x) {
#if __has_builtin(__builtin_amdgcn_tanhf)
  return __builtin_amdgcn_tanhf(x);              // v_tanh_f32 (TRANS32)
#elif __has_builtin(__builtin_amdgcn_tanh_f32)
  return __builtin_amdgcn_tanh_f32(x);
#else
  const float e = __expf(2.0f * x);
#if __has_builtin(__builtin_amdgcn_rcpf)
  return (e - 1.0f) * __builtin_amdgcn_rcpf(e + 1.0f);
#else
  return (e - 1.0f) / (e + 1.0f);
#endif
#endif
}
__device__ __forceinline__ float fast_sigmoid(float x) {
  return 0.5f * fast_tanh(0.5f * x) + 0.5f;
}

__global__ __launch_bounds__(NTH) void biomor_gru_wmma(
    const float* __restrict__ x,    // [B,T,6]
    const float* __restrict__ Wih,  // [384,6]
    const float* __restrict__ Whh,  // [384,128]
    const float* __restrict__ bih,  // [384]
    const float* __restrict__ bhh,  // [384]
    const float* __restrict__ Wg,   // [2,128]
    const float* __restrict__ bg,   // [2]
    const float* __restrict__ Wb,   // [4,128]
    const float* __restrict__ bb,   // [4]
    const float* __restrict__ W1,   // [128,2]
    const float* __restrict__ b1,   // [128]
    const float* __restrict__ W2,   // [4,128]
    const float* __restrict__ b2,   // [4]
    float* __restrict__ y_out,      // [B,T,4]
    float* __restrict__ rout)       // [B,T,128]
{
  extern __shared__ float smem[];
  // bf16 blocks first: 16B-aligned, all sizes are 16B multiples
  __bf16* sh_hbf   = (__bf16*)smem;                      // TB*KAUG   (5120 B)
  __bf16* sh_hdnbf = sh_hbf + TB * KAUG;                 // TB*128    (4096 B)
  __bf16* sh_hf0   = sh_hdnbf + TB * HIDDEN;             // 4*32*16   (4096 B) head0 frag [Wg|Wb]
  __bf16* sh_hf1   = sh_hf0 + 4 * 32 * 16;               // 4*32*16   (4096 B) head1 frag [W2]
  float* fbase   = smem + (TB * KAUG + TB * HIDDEN + 2 * 4 * 32 * 16) / 2;
  float* sh_pre  = fbase;                         // TB*HWS : combined pre-activation (also prologue scratch)
  float* sh_xwn  = sh_pre + TB * HWS;             // TB*XWNS: xw_n (incl b_ih) for n-gate
  float* sh_h    = sh_xwn + TB * XWNS;            // TB*128 : f32 master hidden
  float* sh_xs   = sh_h   + TB * HIDDEN;          // TB*8   : x[:,t,:] f32 (padded)
  float* sh_ph   = sh_xs  + TB * 8;               // TB*PHS : head tile (gain0,1,bias0..3 + bg/bb)
  float* sh_rp   = sh_ph  + TB * PHS;             // TB*PHS : reflex tile (incl b2)
  float* sh_Wihn = sh_rp  + TB * PHS;             // 128*8  : W_ih rows 256..383 (padded 6->8)
  float* sh_bihn = sh_Wihn+ HIDDEN * 8;           // 128
  float* sh_W1   = sh_bihn+ HIDDEN;               // 256
  float* sh_b1   = sh_W1  + HIDDEN * 2;           // 128

  const int tid  = threadIdx.x;
  const int lane = tid & 31;
  const int wv   = tid >> 5;         // wave 0..7 owns gate columns [48*wv, 48*wv+48)
  const int wvu  = __builtin_amdgcn_readfirstlane(wv);   // provably wave-uniform
  const int b0   = blockIdx.x * TB;

  const int cn    = lane & 15;
  const int khalf = (lane < 16) ? 0 : 16;
  const int rb    = (lane < 16) ? 0 : 8;   // C layout: M = vgpr idx (+8 for hi half)
  const int alo   = (lane < 16) ? 0 : 16;  // A/B K order in this code: contiguous (16 per half)

  // ---- Prologue: small params into LDS (coalesced) ----
  for (int i = tid; i < HIDDEN * CTXN; i += NTH) {
    const int j = i / CTXN, c = i - j * CTXN;
    sh_Wihn[j * 8 + c] = Wih[2 * HIDDEN * CTXN + i];
  }
  for (int i = tid; i < HIDDEN; i += NTH) { sh_bihn[i] = bih[2 * HIDDEN + i]; sh_b1[i] = b1[i]; }
  for (int i = tid; i < 2 * HIDDEN; i += NTH) sh_W1[i] = W1[i];
  for (int i = tid; i < TB * HIDDEN; i += NTH) { sh_h[i] = 0.0f; sh_hdnbf[i] = (__bf16)0.0f; }
  for (int i = tid; i < TB * KAUG; i += NTH) sh_hbf[i] = (__bf16)0.0f;

  // Stage head weights in sh_pre scratch (coalesced), then pack LDS-resident fragments
  {
    float* tmpW = sh_pre;  // rows 0-1: Wg, rows 2-5: Wb, rows 6-9: W2
    for (int i = tid; i < 2 * HIDDEN; i += NTH) tmpW[i] = Wg[i];
    for (int i = tid; i < 4 * HIDDEN; i += NTH) tmpW[2 * HIDDEN + i] = Wb[i];
    for (int i = tid; i < 4 * HIDDEN; i += NTH) tmpW[6 * HIDDEN + i] = W2[i];
    __syncthreads();
    // fragment layout: [kc][lane][e] , element K = kc*32 + (lane<16?0:16) + e, col = lane&15
    for (int idx = tid; idx < 2 * 2048; idx += NTH) {
      const int buf = idx >> 11;
      const int r   = idx & 2047;
      const int kc  = r >> 9;
      const int ln  = (r >> 4) & 31;
      const int e   = r & 15;
      const int k   = kc * 32 + ((ln < 16) ? 0 : 16) + e;
      const int c2  = ln & 15;
      float v = 0.0f;
      if (buf == 0) { if (c2 < 6) v = tmpW[c2 * HIDDEN + k]; }
      else          { if (c2 < 4) v = tmpW[(6 + c2) * HIDDEN + k]; }
      (buf == 0 ? sh_hf0 : sh_hf1)[r] = (__bf16)v;
    }
  }

  // Per-lane head C-init (scalar, registers)
  float hcinit = 0.0f;
  if (wvu == 0)      hcinit = (cn < 2) ? bg[cn] : (cn < 6 ? bb[cn - 2] : 0.0f);
  else if (wvu == 1) hcinit = (cn < 4) ? b2[cn] : 0.0f;

  // ---- Register-resident recurrence B-fragments (augmented 160x16 per tile):
  //      K<128 from W_hh (float4 loads), K in [128,134) from W_ih, rest 0.
  v16bf Bfrag[3][5];
  float cinit[3];
#pragma unroll
  for (int tt = 0; tt < 3; ++tt) {
    const int g = wv * 48 + tt * 16 + cn;   // gate column (row of Whh/Wih)
#pragma unroll
    for (int kc = 0; kc < 4; ++kc) {
      const v4f* src4 = (const v4f*)(Whh + (size_t)g * HIDDEN + kc * 32 + khalf);
      const v4f q0 = src4[0], q1 = src4[1], q2 = src4[2], q3 = src4[3];
      v16bf bf;
#pragma unroll
      for (int e = 0; e < 4; ++e) {
        bf[e]      = (__bf16)q0[e];
        bf[4 + e]  = (__bf16)q1[e];
        bf[8 + e]  = (__bf16)q2[e];
        bf[12 + e] = (__bf16)q3[e];
      }
      Bfrag[tt][kc] = bf;
    }
    { // augmented chunk kc=4: K = 128..159 (x columns), zeros for lanes>=16
      v16bf bf;
#pragma unroll
      for (int e = 0; e < 16; ++e) {
        const int K = 128 + khalf + e;
        bf[e] = (K < 128 + CTXN) ? (__bf16)Wih[(size_t)g * CTXN + (K - 128)] : (__bf16)0.0f;
      }
      Bfrag[tt][4] = bf;
    }
    cinit[tt] = bhh[g] + bih[g];
  }
  __syncthreads();

  // Running global pointers (advance by constants per step; deref is guarded)
  const int xrow = tid / CTXN, xi = tid - xrow * CTXN;
  const float* xp = x + ((size_t)(b0 + xrow) * TSTEPS) * CTXN + xi;
  float* rp = rout + ((size_t)(b0 + (tid >> 4)) * TSTEPS) * HIDDEN + (tid & 15);
  float* yp = y_out + ((size_t)(b0 + (tid >> 2)) * TSTEPS) * OUTN + (tid & 3);

  // ---- Recurrence over T ----
  for (int t = 0; t < TSTEPS; ++t) {
    // Stage x[:, t, :] : f32 copy for reflex head, bf16 into augmented A rows
    if (tid < TB * CTXN) {
      const float v = *xp;
      sh_xs[xrow * 8 + xi] = v;
      sh_hbf[xrow * KAUG + HIDDEN + xi] = (__bf16)v;
    }
    __syncthreads();

    // xw_n (n-gate input projection) in VALU, needed separately for r*hw_n
#pragma unroll
    for (int u = 0; u < (TB * HIDDEN) / NTH; ++u) {
      const int flat = tid + u * NTH;
      const int row = flat >> 7, j = flat & (HIDDEN - 1);
      const float* wr = sh_Wihn + j * 8;
      const float* xr = sh_xs + row * 8;
      float acc = sh_bihn[j];
#pragma unroll
      for (int i = 0; i < CTXN; ++i) acc += xr[i] * wr[i];
      sh_xwn[row * XWNS + j] = acc;
    }

    // Combined pre-activation: [h|x] @ [Whh|Wih]^T + (bhh+bih), K=160 -> 5 WMMA/tile
    {
      const __bf16* hb = sh_hbf + cn * KAUG;
      v16bf afr[5];
#pragma unroll
      for (int kc = 0; kc < 5; ++kc) {
        const int k0 = kc * 32 + alo;
        const v8bf lo = *(const v8bf*)(hb + k0);        // ds_load_b128
        const v8bf hi = *(const v8bf*)(hb + k0 + 8);
        afr[kc] = __builtin_shufflevector(lo, hi, 0,1,2,3,4,5,6,7,8,9,10,11,12,13,14,15);
      }
#pragma unroll
      for (int tt = 0; tt < 3; ++tt) {
        const int colg = wv * 48 + tt * 16 + cn;
        const float bh = cinit[tt];
        v8f c = { bh, bh, bh, bh, bh, bh, bh, bh };
#pragma unroll
        for (int kc = 0; kc < 5; ++kc) {
          c = __builtin_amdgcn_wmma_f32_16x16x32_bf16(
              false, afr[kc], false, Bfrag[tt][kc], (short)0, c, false, false);
        }
#pragma unroll
        for (int i = 0; i < 8; ++i) sh_pre[(rb + i) * HWS + colg] = c[i];
      }
    }
    __syncthreads();

    // GRU gate update: r,z from combined; n recovers hw_n = pre_n - xw_n
    {
      const int row = tid >> 4, jb = tid & 15;
      float* hrow = sh_h + row * HIDDEN;
      const float* prer = sh_pre + row * HWS;
      const float* xwnr = sh_xwn + row * XWNS;
#pragma unroll
      for (int u = 0; u < 8; ++u) {
        const int j = jb + u * 16;
        const float r   = fast_sigmoid(prer[j]);
        const float z   = fast_sigmoid(prer[j + HIDDEN]);
        const float xn  = xwnr[j];
        const float hwn = prer[j + 2 * HIDDEN] - xn;
        const float n   = fast_tanh(xn + r * hwn);
        const float hn  = (1.0f - z) * n + z * hrow[j];
        hrow[j] = hn;
        sh_hbf[row * KAUG + j] = (__bf16)hn;
        rp[u * 16] = hn;
      }
    }
    __syncthreads();

    // Head GEMM (wave 0): h_new @ [Wg|Wb]^T + [bg|bb] -> sh_ph  (4 WMMA, B from LDS)
    if (wvu == 0) {
      const __bf16* hb = sh_hbf + cn * KAUG;
      v8f c = { hcinit, hcinit, hcinit, hcinit, hcinit, hcinit, hcinit, hcinit };
#pragma unroll
      for (int kc = 0; kc < 4; ++kc) {
        const int k0 = kc * 32 + alo;
        const v8bf alov = *(const v8bf*)(hb + k0);
        const v8bf ahiv = *(const v8bf*)(hb + k0 + 8);
        const v16bf a = __builtin_shufflevector(alov, ahiv, 0,1,2,3,4,5,6,7,8,9,10,11,12,13,14,15);
        const __bf16* fb = sh_hf0 + (kc * 32 + lane) * 16;
        const v8bf blo = *(const v8bf*)fb;
        const v8bf bhi = *(const v8bf*)(fb + 8);
        const v16bf bfr = __builtin_shufflevector(blo, bhi, 0,1,2,3,4,5,6,7,8,9,10,11,12,13,14,15);
        c = __builtin_amdgcn_wmma_f32_16x16x32_bf16(false, a, false, bfr, (short)0, c, false, false);
      }
#pragma unroll
      for (int i = 0; i < 8; ++i) sh_ph[(rb + i) * PHS + cn] = c[i];
    }
    __syncthreads();

    // Reflex hidden (K=2, VALU): gain from head tile, hdn -> bf16 A buffer
    {
      const int row = tid >> 4, jb = tid & 15;
      const float g0 = fast_sigmoid(sh_ph[row * PHS + 0]);
      const float g1 = fast_sigmoid(sh_ph[row * PHS + 1]);
      const float m0 = sh_xs[row * 8 + 0] * g0;
      const float m1 = sh_xs[row * 8 + 1] * g1;
#pragma unroll
      for (int u = 0; u < 8; ++u) {
        const int j = jb + u * 16;
        const float v = m0 * sh_W1[j * 2 + 0] + m1 * sh_W1[j * 2 + 1] + sh_b1[j];
        sh_hdnbf[row * HIDDEN + j] = (__bf16)(v > 0.0f ? v : 0.0f);
      }
    }
    __syncthreads();

    // Reflex GEMM (wave 1): hdn @ W2^T + b2 -> sh_rp  (4 WMMA, B from LDS)
    if (wvu == 1) {
      const __bf16* hb = sh_hdnbf + cn * HIDDEN;
      v8f c = { hcinit, hcinit, hcinit, hcinit, hcinit, hcinit, hcinit, hcinit };
#pragma unroll
      for (int kc = 0; kc < 4; ++kc) {
        const int k0 = kc * 32 + alo;
        const v8bf alov = *(const v8bf*)(hb + k0);
        const v8bf ahiv = *(const v8bf*)(hb + k0 + 8);
        const v16bf a = __builtin_shufflevector(alov, ahiv, 0,1,2,3,4,5,6,7,8,9,10,11,12,13,14,15);
        const __bf16* fb = sh_hf1 + (kc * 32 + lane) * 16;
        const v8bf blo = *(const v8bf*)fb;
        const v8bf bhi = *(const v8bf*)(fb + 8);
        const v16bf bfr = __builtin_shufflevector(blo, bhi, 0,1,2,3,4,5,6,7,8,9,10,11,12,13,14,15);
        c = __builtin_amdgcn_wmma_f32_16x16x32_bf16(false, a, false, bfr, (short)0, c, false, false);
      }
#pragma unroll
      for (int i = 0; i < 8; ++i) sh_rp[(rb + i) * PHS + cn] = c[i];
    }
    __syncthreads();

    // Final combine: logit = reflex + bias; sigmoid for 0..1, tanh for 2..3
    if (tid < TB * OUTN) {
      const int row = tid >> 2, o = tid & 3;
      const float acc = sh_rp[row * PHS + o] + sh_ph[row * PHS + 2 + o];
      *yp = (o < 2) ? fast_sigmoid(acc) : fast_tanh(acc);
    }
    __syncthreads();   // protects sh_xs / sh_ph / sh_rp / sh_hbf reuse next step

    xp += CTXN; rp += HIDDEN; yp += OUTN;
  }
}

extern "C" void kernel_launch(void* const* d_in, const int* in_sizes, int n_in,
                              void* d_out, int out_size, void* d_ws, size_t ws_size,
                              hipStream_t stream) {
  const float* x   = (const float*)d_in[0];
  const float* Wih = (const float*)d_in[1];
  const float* Whh = (const float*)d_in[2];
  const float* bih = (const float*)d_in[3];
  const float* bhh = (const float*)d_in[4];
  const float* Wg  = (const float*)d_in[5];
  const float* bg  = (const float*)d_in[6];
  const float* Wb  = (const float*)d_in[7];
  const float* bb  = (const float*)d_in[8];
  const float* W1  = (const float*)d_in[9];
  const float* b1  = (const float*)d_in[10];
  const float* W2  = (const float*)d_in[11];
  const float* b2  = (const float*)d_in[12];

  const int Btot = in_sizes[0] / (TSTEPS * CTXN);   // 2048
  float* y    = (float*)d_out;                       // [B,T,4] first (return order)
  float* rout = y + (size_t)Btot * TSTEPS * OUTN;    // then [B,T,128]

  // Dynamic LDS: bf16 blocks + float region
  size_t smemBytes = ((size_t)TB * KAUG + (size_t)TB * HIDDEN + 2 * 4 * 32 * 16) * 2;
  size_t smemf = (size_t)TB * HWS + (size_t)TB * XWNS + (size_t)TB * HIDDEN
               + TB * 8 + (size_t)TB * PHS * 2
               + (size_t)HIDDEN * 8 + HIDDEN + HIDDEN * 2 + HIDDEN;
  smemBytes += smemf * sizeof(float);

  dim3 grid(Btot / TB), block(NTH);
  hipLaunchKernelGGL(biomor_gru_wmma, grid, block, smemBytes, stream,
                     x, Wih, Whh, bih, bhh, Wg, bg, Wb, bb, W1, b1, W2, b2, y, rout);
}
